// MOSDecoder_36146444763734
// MI455X (gfx1250) — compile-verified, matrix operand order checked by reading, not run on hardware
//
#include <hip/hip_runtime.h>
#include <hip/hip_bf16.h>

// Problem constants (from reference): D=512, E=4, V=32000, B=1, L=1024
#define DD 512
#define EE 4
#define VV 32000
#define LL 1024
#define MROWS (LL * EE)    // 4096 (token,expert) rows

// Big-GEMM blocking: 8 waves/block, each wave one M=16 tile (block M=128),
// all waves share an N=128 strip staged in LDS (8 n-tiles per wave).
#define NTILES 8
#define BN (NTILES * 16)          // 128
#define BM 128
#define LDS_STRIDE 520            // 512 + 8 bf16 pad -> bank-conflict-free b128 frags

typedef __attribute__((ext_vector_type(16))) __bf16 bf16x16;
typedef __attribute__((ext_vector_type(8)))  __bf16 bf16x8;
typedef __attribute__((ext_vector_type(8)))  float  f32x8;

union FragU { bf16x16 v; bf16x8 h[2]; };

// 16-bit WMMA A/B fragment load (wave32).
// ISA layout: lane(0-15) = row, elements 0..7 = K[k0 + half*8 .. +7],
//             elements 8..15 = K[k0+16+half*8 .. +7]; half = lane>>4.
static __device__ inline bf16x16 load_frag(const __bf16* __restrict__ row, int k0, int half) {
  FragU u;
  u.h[0] = *reinterpret_cast<const bf16x8*>(row + k0 + half * 8);
  u.h[1] = *reinterpret_cast<const bf16x8*>(row + k0 + 16 + half * 8);
  return u.v;
}

// CDNA5 async copy: 16 bytes global -> LDS, tracked by ASYNCcnt (no VGPR staging).
// LDS address from the first VGPR operand; global address from the 64-bit vaddr.
static __device__ inline void async_copy_b128(unsigned lds_off, const void* gaddr) {
  asm volatile("global_load_async_to_lds_b128 %0, %1, off"
               :: "v"(lds_off), "v"(gaddr)
               : "memory");
}
static __device__ inline void wait_async0() {
  asm volatile("s_wait_asynccnt 0x0" ::: "memory");
}

// ---------------- prep kernels ----------------

// hidden_states f32 [L][D] -> bf16 [L][D]
__global__ __launch_bounds__(256) void k_conv_hs(const float* __restrict__ hs,
                                                 __bf16* __restrict__ hsb) {
  int i = blockIdx.x * 256 + threadIdx.x;       // < L*D = 524288
  hsb[i] = (__bf16)hs[i];
}

// W_latent f32 [D][E*D] -> bf16 transposed [E*D][D]
__global__ __launch_bounds__(256) void k_conv_wlatT(const float* __restrict__ wl,
                                                    __bf16* __restrict__ wlT) {
  int i = blockIdx.x * 256 + threadIdx.x;       // < D*E*D = 1048576
  int k = i >> 11;                              // row in [0,512)
  int n = i & 2047;                             // col in [0,2048)
  wlT[(size_t)n * DD + k] = (__bf16)wl[i];
}

// W_dec f32 [D][V] -> bf16 transposed [V][D], LDS-tiled 32x32
__global__ __launch_bounds__(256) void k_conv_wdecT(const float* __restrict__ wd,
                                                    __bf16* __restrict__ wdT) {
  __shared__ float tile[32][33];
  int vBase = blockIdx.x * 32;                  // grid.x = 1000
  int kBase = blockIdx.y * 32;                  // grid.y = 16
  int tx = threadIdx.x & 31;
  int ty = threadIdx.x >> 5;                    // 0..7
  #pragma unroll
  for (int i = 0; i < 32; i += 8)
    tile[ty + i][tx] = wd[(size_t)(kBase + ty + i) * VV + vBase + tx];
  __syncthreads();
  #pragma unroll
  for (int i = 0; i < 32; i += 8)
    wdT[(size_t)(vBase + ty + i) * DD + kBase + tx] = (__bf16)tile[tx][ty + i];
}

// mixture prior: log_softmax(hs @ W_prior + b_prior) -> log_coef [L*E] (flat row = l*4+e)
__global__ __launch_bounds__(256) void k_prior(const float* __restrict__ hs,
                                               const float* __restrict__ Wp,
                                               const float* __restrict__ bp,
                                               float* __restrict__ log_coef) {
  int l = blockIdx.x * 256 + threadIdx.x;
  if (l >= LL) return;
  const float* h = hs + (size_t)l * DD;
  float acc[EE] = {bp[0], bp[1], bp[2], bp[3]};
  for (int k = 0; k < DD; ++k) {
    float x = h[k];
    #pragma unroll
    for (int e = 0; e < EE; ++e) acc[e] += x * Wp[k * EE + e];
  }
  float m = fmaxf(fmaxf(acc[0], acc[1]), fmaxf(acc[2], acc[3]));
  float s = 0.f;
  #pragma unroll
  for (int e = 0; e < EE; ++e) s += expf(acc[e] - m);
  float lse = m + logf(s);
  #pragma unroll
  for (int e = 0; e < EE; ++e) log_coef[l * EE + e] = acc[e] - lse;
}

// ---------------- latent projection: tanh(hs @ W_latent + b) via WMMA ----------------
// Output stored as bf16 rows latent[(l*4+e)][d] so the big GEMM's A-fragments
// are plain K-major row loads.
__global__ __launch_bounds__(256) void k_latent(const __bf16* __restrict__ hsb,
                                                const __bf16* __restrict__ wlT,
                                                const float* __restrict__ bl,
                                                __bf16* __restrict__ latent) {
  int lane = threadIdx.x & 31, wave = threadIdx.x >> 5;
  int laneN = lane & 15, half = lane >> 4;
  int m0  = blockIdx.x * 16;                    // token tile base, grid.x = 64
  int n0w = blockIdx.y * 512 + wave * 64;       // grid.y = 4 -> covers 2048 cols
  const __bf16* arow = hsb + (size_t)(m0 + laneN) * DD;
  const __bf16* brow[4];
  #pragma unroll
  for (int t = 0; t < 4; ++t) brow[t] = wlT + (size_t)(n0w + t * 16 + laneN) * DD;
  f32x8 acc[4] = {};
  for (int k0 = 0; k0 < DD; k0 += 32) {
    bf16x16 a = load_frag(arow, k0, half);
    #pragma unroll
    for (int t = 0; t < 4; ++t) {
      bf16x16 b = load_frag(brow[t], k0, half);
      acc[t] = __builtin_amdgcn_wmma_f32_16x16x32_bf16(false, a, false, b,
                                                       (short)0, acc[t], false, false);
    }
  }
  #pragma unroll
  for (int t = 0; t < 4; ++t) {
    int n = n0w + t * 16 + laneN;
    float bn = bl[n];
    #pragma unroll
    for (int r = 0; r < 8; ++r) {
      int tok = m0 + half * 8 + r;
      float val = tanhf(acc[t][r] + bn);
      latent[(size_t)(tok * EE + (n >> 9)) * DD + (n & (DD - 1))] = (__bf16)val;
    }
  }
}

// ---------------- shared big-GEMM core ----------------
// Block: 8 waves; wave w computes rows [blockIdx.x*128 + w*16, +16) over the
// block's shared N=128 strip (staged in LDS via async copy). acc[t] = 16x16 f32 tile.
static __device__ inline void gemm_block(const __bf16* __restrict__ latent,
                                         const __bf16* __restrict__ wdT,
                                         __bf16* __restrict__ Bs,
                                         f32x8 (&acc)[NTILES],
                                         int& m0w, int& n0b) {
  int lane = threadIdx.x & 31, wave = threadIdx.x >> 5;
  int laneN = lane & 15, half = lane >> 4;
  m0w = blockIdx.x * BM + wave * 16;
  n0b = blockIdx.y * BN;

  // cooperative fill: 128 rows x 512 K bf16 -> LDS (rows padded to 520),
  // using CDNA5 async global->LDS b128 copies (ASYNCcnt), no VGPR staging.
  for (int c = threadIdx.x; c < BN * (DD / 8); c += 256) {
    int row = c >> 6;                 // 0..127
    int ch  = c & 63;                 // 16B chunk within row
    unsigned lds_off = (unsigned)(uintptr_t)(Bs + row * LDS_STRIDE + ch * 8);
    async_copy_b128(lds_off, wdT + (size_t)(n0b + row) * DD + ch * 8);
  }
  wait_async0();                      // our async writes to LDS are done
  __syncthreads();                    // everyone's writes visible block-wide

  const __bf16* arow = latent + (size_t)(m0w + laneN) * DD;
  const __bf16* brow[NTILES];
  #pragma unroll
  for (int t = 0; t < NTILES; ++t) brow[t] = Bs + (t * 16 + laneN) * LDS_STRIDE;

  #pragma unroll
  for (int t = 0; t < NTILES; ++t) acc[t] = (f32x8){};

  for (int k0 = 0; k0 < DD; k0 += 32) {
    bf16x16 a = load_frag(arow, k0, half);
    #pragma unroll
    for (int t = 0; t < NTILES; ++t) {
      bf16x16 b = load_frag(brow[t], k0, half);
      acc[t] = __builtin_amdgcn_wmma_f32_16x16x32_bf16(false, a, false, b,
                                                       (short)0, acc[t], false, false);
    }
  }
}

// ---------------- pass 1: Z[row] = sum_v exp(logit[row][v] + b_dec[v]) ----------------
__global__ __launch_bounds__(256) void k_pass1(const __bf16* __restrict__ latent,
                                               const __bf16* __restrict__ wdT,
                                               const float* __restrict__ bd,
                                               float* __restrict__ Z) {
  __shared__ __bf16 Bs[BN * LDS_STRIDE];        // 133,120 B (CDNA5: 320KB/WG)
  int lane = threadIdx.x & 31;
  int laneN = lane & 15, half = lane >> 4;
  f32x8 acc[NTILES];
  int m0w, n0b;
  gemm_block(latent, wdT, Bs, acc, m0w, n0b);

  float part[8] = {0, 0, 0, 0, 0, 0, 0, 0};
  #pragma unroll
  for (int t = 0; t < NTILES; ++t) {
    float b = bd[n0b + t * 16 + laneN];
    #pragma unroll
    for (int r = 0; r < 8; ++r) part[r] += __expf(acc[t][r] + b);
  }
  // reduce across the 16 lanes of each half-wave (xor masks stay within half)
  #pragma unroll
  for (int m = 1; m < 16; m <<= 1)
    #pragma unroll
    for (int r = 0; r < 8; ++r) part[r] += __shfl_xor(part[r], m, 32);
  if (laneN == 0) {
    #pragma unroll
    for (int r = 0; r < 8; ++r) atomicAdd(&Z[m0w + half * 8 + r], part[r]);
  }
}

// ---------------- pass 2: out[l][v] = log( sum_e exp(logit)*coef_e/Z_e ) ----------------
__global__ __launch_bounds__(256) void k_pass2(const __bf16* __restrict__ latent,
                                               const __bf16* __restrict__ wdT,
                                               const float* __restrict__ bd,
                                               const float* __restrict__ Z,
                                               const float* __restrict__ log_coef,
                                               float* __restrict__ out) {
  __shared__ __bf16 Bs[BN * LDS_STRIDE];
  int lane = threadIdx.x & 31;
  int laneN = lane & 15, half = lane >> 4;
  f32x8 acc[NTILES];
  int m0w, n0b;
  gemm_block(latent, wdT, Bs, acc, m0w, n0b);

  // per-row weight coef_e / Z_e ; row = m0w + half*8 + r = token*4 + e
  float Cw[8];
  #pragma unroll
  for (int r = 0; r < 8; ++r) {
    int row = m0w + half * 8 + r;
    Cw[r] = __expf(log_coef[row]) / Z[row];
  }
  int tA = (m0w >> 2) + half * 2;               // rows r=0..3 -> token tA, r=4..7 -> tA+1
  #pragma unroll
  for (int t = 0; t < NTILES; ++t) {
    int v = n0b + t * 16 + laneN;
    float b = bd[v];
    float sA = 0.f, sB = 0.f;
    #pragma unroll
    for (int r = 0; r < 4; ++r) {
      sA += Cw[r]     * __expf(acc[t][r]     + b);
      sB += Cw[r + 4] * __expf(acc[t][r + 4] + b);
    }
    out[(size_t)tA * VV + v]       = __logf(sA);
    out[(size_t)(tA + 1) * VV + v] = __logf(sB);
  }
}

// ---------------- host launch ----------------

extern "C" void kernel_launch(void* const* d_in, const int* in_sizes, int n_in,
                              void* d_out, int out_size, void* d_ws, size_t ws_size,
                              hipStream_t stream) {
  const float* hs  = (const float*)d_in[0];   // [1,1024,512]
  const float* Wp  = (const float*)d_in[1];   // [512,4]
  const float* bp  = (const float*)d_in[2];   // [4]
  const float* Wl  = (const float*)d_in[3];   // [512,2048]
  const float* bl  = (const float*)d_in[4];   // [2048]
  const float* Wd  = (const float*)d_in[5];   // [512,32000]
  const float* bd  = (const float*)d_in[6];   // [32000]
  float* out = (float*)d_out;                 // [1,1024,32000]

  char* ws = (char*)d_ws;
  // workspace layout (all 16B-aligned)
  __bf16* hsb     = (__bf16*)(ws + 0);            // 1,048,576 B
  __bf16* wlT     = (__bf16*)(ws + 1048576);      // 2,097,152 B
  __bf16* wdT     = (__bf16*)(ws + 3145728);      // 32,768,000 B
  __bf16* latent  = (__bf16*)(ws + 35913728);     // 4,194,304 B
  float*  logcoef = (float*)(ws + 40108032);      // 16,384 B
  float*  Zrow    = (float*)(ws + 40124416);      // 16,384 B

  hipMemsetAsync(Zrow, 0, MROWS * sizeof(float), stream);

  k_conv_hs   <<<2048, 256, 0, stream>>>(hs, hsb);
  k_conv_wlatT<<<4096, 256, 0, stream>>>(Wl, wlT);
  k_conv_wdecT<<<dim3(VV / 32, DD / 32), 256, 0, stream>>>(Wd, wdT);
  k_prior     <<<(LL + 255) / 256, 256, 0, stream>>>(hs, Wp, bp, logcoef);
  k_latent    <<<dim3(LL / 16, 4),  256, 0, stream>>>(hsb, wlT, bl, latent);
  k_pass1     <<<dim3(MROWS / BM, VV / BN), 256, 0, stream>>>(latent, wdT, bd, Zrow);
  k_pass2     <<<dim3(MROWS / BM, VV / BN), 256, 0, stream>>>(latent, wdT, bd, Zrow, logcoef, out);
}
